// RoformerAttention_23691039605422
// MI455X (gfx1250) — matrix operand order repaired and checked by reading, not compiled, not run on hardware
//
#include <hip/hip_runtime.h>

// ---------------------------------------------------------------- constants
constexpr int Bc  = 2;
constexpr int Sc  = 2048;
constexpr int Dc  = 1024;
constexpr int Hc  = 16;
constexpr int HDc = 64;
constexpr int BS  = Bc * Sc;          // 4096 rows

typedef __attribute__((ext_vector_type(16))) __bf16 v16bf;
typedef __attribute__((ext_vector_type(8)))  float  v8f;

union FragBF { v16bf v; unsigned u[8]; };
union FragF  { v8f   v; float    f[8]; };

__device__ __forceinline__ v8f wmma_bf16(v16bf a, v16bf b, v8f c) {
  // D = A(16x32 bf16) * B(32x16 bf16) + C(16x16 f32)
  return __builtin_amdgcn_wmma_f32_16x16x32_bf16(false, a, false, b,
                                                 (short)0, c, false, false);
}

// round-to-nearest-even f32 -> bf16 (as raw u16)
__device__ __forceinline__ unsigned short f2bf(float f) {
  unsigned u; __builtin_memcpy(&u, &f, 4);
  u += 0x7FFFu + ((u >> 16) & 1u);
  return (unsigned short)(u >> 16);
}
__device__ __forceinline__ unsigned pack_bf16(float a, float b) {
  return (unsigned)f2bf(a) | ((unsigned)f2bf(b) << 16);
}

// Async global->LDS copy of 32 contiguous bytes (two B128 ops).
// dsaddr = lds_off + INST_OFFSET ; memaddr = gaddr + INST_OFFSET (both advance).
// Requires 16B alignment on both addresses. Tracked by ASYNCcnt.
__device__ __forceinline__ void async_copy32(unsigned lds_off, const void* gaddr) {
  asm volatile("global_load_async_to_lds_b128 %0, %1, off\n\t"
               "global_load_async_to_lds_b128 %0, %1, off offset:16"
               :: "v"(lds_off), "v"(gaddr) : "memory");
}
__device__ __forceinline__ void wait_async0() {
  asm volatile("s_wait_asynccnt 0x0" ::: "memory");
}
__device__ __forceinline__ unsigned lds_off_u32(const void* p) {
  // generic pointers to LDS carry the byte offset in the low 32 bits
  return (unsigned)(size_t)p;
}

// A fragment (16x32, M x K): row pointer supplied by caller.
// u32 words: j=0..3 -> K=8h+2j ; j=4..7 -> K=16+8h+2(j-4)
__device__ __forceinline__ v16bf ldsA_frag(const unsigned short* row, int half) {
  FragBF a; const unsigned* p = (const unsigned*)row;
#pragma unroll
  for (int j = 0; j < 4; ++j) { a.u[j] = p[4*half + j]; a.u[4+j] = p[8 + 4*half + j]; }
  return a.v;
}
// B fragment (32x16, K x N) from LDS stored as [n][k]: words j=0..3 -> K=8h+2j ; j=4..7 -> K=16+8h+2(j-4)
__device__ __forceinline__ v16bf ldsB_frag(const unsigned short* rowN, int half) {
  FragBF b; const unsigned* p = (const unsigned*)rowN;
#pragma unroll
  for (int j = 0; j < 8; ++j) b.u[j] = p[4*half + j + (j & 4)];
  return b.v;
}

// ================================================================ QKV proj
// out[m, n] = sum_k X[m,k] * W[n,k]  (+bias, +optional RoPE), stored bf16.
__global__ __launch_bounds__(128)
void proj_kernel(const float* __restrict__ X, const float* __restrict__ W,
                 const float* __restrict__ bias, unsigned short* __restrict__ out,
                 int apply_rope)
{
  __shared__ unsigned short lA[64][34];   // X tile  [m][k], padded
  __shared__ unsigned short lB[64][34];   // W tile  [n][k], padded

  const int tid  = threadIdx.x;
  const int wave = tid >> 5;
  const int lane = tid & 31;
  const int half = lane >> 4;
  const int ln   = lane & 15;
  const int mBase = blockIdx.x * 64;
  const int nBase = blockIdx.y * 64;

  v8f acc[4];
#pragma unroll
  for (int t = 0; t < 4; ++t) { FragF z; for (int r = 0; r < 8; ++r) z.f[r] = 0.f; acc[t] = z.v; }

  const int lr = tid >> 1;            // 0..63 : tile row handled by this thread
  const int lc = (tid & 1) * 16;      // 0/16  : 16-col half

  for (int k0 = 0; k0 < Dc; k0 += 32) {
    __syncthreads();
    {
      const float* sa = X + (size_t)(mBase + lr) * Dc + k0 + lc;
      const float* sb = W + (size_t)(nBase + lr) * Dc + k0 + lc;
      unsigned* da = (unsigned*)&lA[lr][lc];
      unsigned* db = (unsigned*)&lB[lr][lc];
#pragma unroll
      for (int j = 0; j < 8; ++j) da[j] = pack_bf16(sa[2*j], sa[2*j+1]);
#pragma unroll
      for (int j = 0; j < 8; ++j) db[j] = pack_bf16(sb[2*j], sb[2*j+1]);
    }
    __syncthreads();

    v16bf afrag = ldsA_frag(&lA[wave*16 + ln][0], half);
#pragma unroll
    for (int t = 0; t < 4; ++t) {
      v16bf bfrag = ldsB_frag(&lB[t*16 + ln][0], half);
      acc[t] = wmma_bf16(afrag, bfrag, acc[t]);
    }
  }

  // epilogue: bias (+RoPE), cast bf16, store
#pragma unroll
  for (int t = 0; t < 4; ++t) {
    FragF cv; cv.v = acc[t];
    const int col  = nBase + t*16 + ln;
    const float bv = bias[col];
#pragma unroll
    for (int r = 0; r < 8; ++r) {
      const int row = mBase + wave*16 + half*8 + r;
      float v = cv.f[r] + bv;
      if (apply_rope) {
        const int s = row & (Sc - 1);
        const int j = col & (HDc - 1);
        const int i = j & 31;
        const float freq = __expf(-0.28782313662425572f * (float)i); // 10000^(-i/32)
        const float ang  = (float)s * freq;
        const float cs = __builtin_cosf(ang);
        const float sn = __builtin_sinf(ang);
        const float partner = __shfl_xor(v, 1);
        const float rot = (j & 1) ? partner : -partner;   // rot[2i]=-x[2i+1], rot[2i+1]=x[2i]
        v = v * cs + rot * sn;
      }
      out[(size_t)row * Dc + col] = f2bf(v);
    }
  }
}

// ================================================================ attention
// Flash-style: block = (64 queries) x one (b,h); waves own 16-query tiles.
__global__ __launch_bounds__(128)
void attn_kernel(const unsigned short* __restrict__ Qb,
                 const unsigned short* __restrict__ Kb,
                 const unsigned short* __restrict__ Vb,
                 const float* __restrict__ mask,   // (B,1,S,S)
                 const float* __restrict__ cdrw,   // scalar
                 unsigned short* __restrict__ Ob)  // (BS, D) bf16
{
  __shared__ unsigned short lK[32][72];      // [key][d], 144B rows (16B-aligned for async b128)
  __shared__ unsigned short lVt[64][34];     // [hd][key]   (V transposed)
  __shared__ unsigned short lP[4][16][34];   // per-wave P staging [m][key]

  const int tid  = threadIdx.x;
  const int wave = tid >> 5;
  const int lane = tid & 31;
  const int half = lane >> 4;
  const int ln   = lane & 15;

  const int h = blockIdx.y;
  const int b = blockIdx.z;
  const float  cw      = cdrw[0];
  const size_t headOff = (size_t)h * HDc;
  const size_t rowBase = (size_t)b * Sc;
  const int    qRow    = blockIdx.x * 64 + wave * 16;

  // -------- Q fragments (2 K-chunks of 32), loaded once (A layout)
  v16bf aq0, aq1;
  {
    const unsigned* p = (const unsigned*)(Qb + (rowBase + qRow + ln) * (size_t)Dc + headOff);
    FragBF a0, a1;
#pragma unroll
    for (int j = 0; j < 4; ++j) { a0.u[j] = p[4*half + j];      a0.u[4+j] = p[8  + 4*half + j]; }
#pragma unroll
    for (int j = 0; j < 4; ++j) { a1.u[j] = p[16 + 4*half + j]; a1.u[4+j] = p[24 + 4*half + j]; }
    aq0 = a0.v; aq1 = a1.v;
  }

  float mrow[8], lrow[8];
  v8f acc[4];
#pragma unroll
  for (int r = 0; r < 8; ++r) { mrow[r] = -3.0e38f; lrow[r] = 0.f; }
#pragma unroll
  for (int t = 0; t < 4; ++t) { FragF z; for (int r = 0; r < 8; ++r) z.f[r] = 0.f; acc[t] = z.v; }

  // per-thread staging coordinates
  const int kr = tid >> 2;              // 0..31 : key row
  const int kc = (tid & 3) * 16;        // 0,16,32,48 : 16-elem column group

  for (int kt = 0; kt < Sc; kt += 32) {
    __syncthreads();                    // previous-iter LDS reads done
    {                                   // stage K async (CDNA5 async-LDS path), V via transpose
      const unsigned short* ks = Kb + (rowBase + kt + kr) * (size_t)Dc + headOff + kc;
      async_copy32(lds_off_u32(&lK[kr][kc]), ks);                 // ASYNCcnt-tracked
      if (kt + 32 < Sc)                                           // prefetch next K tile
        __builtin_prefetch((const void*)(ks + (size_t)32 * Dc), 0, 0);
      const unsigned short* vs = Vb + (rowBase + kt + kr) * (size_t)Dc + headOff + kc;
#pragma unroll
      for (int j = 0; j < 16; ++j) lVt[kc + j][kr] = vs[j];
    }
    wait_async0();                      // own wave's async LDS writes landed
    __syncthreads();                    // tiles visible block-wide

    // -------- S = Q K^T * scale + mask*cw ; online softmax
    float alpha[8];
    {
      v8f s0, s1;
      { FragF z; for (int r = 0; r < 8; ++r) z.f[r] = 0.f; s0 = z.v; s1 = z.v; }
      {
        const unsigned* p = (const unsigned*)&lK[ln][0];         // key half 0
        FragBF b0, b1;
#pragma unroll
        for (int j = 0; j < 8; ++j) { b0.u[j] = p[4*half + j + (j & 4)]; b1.u[j] = p[16 + 4*half + j + (j & 4)]; }
        s0 = wmma_bf16(aq0, b0.v, s0);
        s0 = wmma_bf16(aq1, b1.v, s0);
      }
      {
        const unsigned* p = (const unsigned*)&lK[16 + ln][0];    // key half 1
        FragBF b0, b1;
#pragma unroll
        for (int j = 0; j < 8; ++j) { b0.u[j] = p[4*half + j + (j & 4)]; b1.u[j] = p[16 + 4*half + j + (j & 4)]; }
        s1 = wmma_bf16(aq0, b0.v, s1);
        s1 = wmma_bf16(aq1, b1.v, s1);
      }

      FragF sv0, sv1; sv0.v = s0; sv1.v = s1;
      const float scale = 0.125f;           // 1/sqrt(64)
#pragma unroll
      for (int r = 0; r < 8; ++r) {
        const int q = qRow + half*8 + r;
        const float* mq = mask + ((size_t)b * Sc + q) * Sc + kt;
        const float v0 = sv0.f[r] * scale + mq[ln]      * cw;
        const float v1 = sv1.f[r] * scale + mq[16 + ln] * cw;
        float tmax = fmaxf(v0, v1);
#pragma unroll
        for (int off = 1; off < 16; off <<= 1) tmax = fmaxf(tmax, __shfl_xor(tmax, off));
        const float mnew = fmaxf(mrow[r], tmax);
        alpha[r] = __expf(mrow[r] - mnew);
        mrow[r]  = mnew;
        const float e0 = __expf(v0 - mnew);
        const float e1 = __expf(v1 - mnew);
        float tsum = e0 + e1;
#pragma unroll
        for (int off = 1; off < 16; off <<= 1) tsum += __shfl_xor(tsum, off);
        lrow[r] = lrow[r] * alpha[r] + tsum;
        lP[wave][half*8 + r][ln]      = f2bf(e0);
        lP[wave][half*8 + r][16 + ln] = f2bf(e1);
      }
      // rescale O accumulators
#pragma unroll
      for (int t = 0; t < 4; ++t) {
        FragF a_; a_.v = acc[t];
#pragma unroll
        for (int r = 0; r < 8; ++r) a_.f[r] *= alpha[r];
        acc[t] = a_.v;
      }
    }
    __syncthreads();                        // P visible

    // -------- O += P V   (A = P 16x32, B = V 32x16 per hd tile)
    {
      const unsigned* pp = (const unsigned*)&lP[wave][ln][0];
      FragBF ap;
#pragma unroll
      for (int j = 0; j < 4; ++j) { ap.u[j] = pp[4*half + j]; ap.u[4+j] = pp[8 + 4*half + j]; }
#pragma unroll
      for (int t = 0; t < 4; ++t) {
        const unsigned* vp = (const unsigned*)&lVt[t*16 + ln][0];
        FragBF bv;
#pragma unroll
        for (int j = 0; j < 8; ++j) bv.u[j] = vp[4*half + j + (j & 4)];
        acc[t] = wmma_bf16(ap.v, bv.v, acc[t]);
      }
    }
  }

  // -------- normalize and store O (bf16, layout (b,s,h,hd) == (BS, D))
#pragma unroll
  for (int t = 0; t < 4; ++t) {
    FragF a_; a_.v = acc[t];
#pragma unroll
    for (int r = 0; r < 8; ++r) {
      const int q = qRow + half*8 + r;
      Ob[(rowBase + q) * (size_t)Dc + headOff + t*16 + ln] = f2bf(a_.f[r] / lrow[r]);
    }
  }
}

// ================================================================ out-proj
// Y[m,n] = sum_k A[m,k]*Wo[n,k] + bo[n] + X[m,n]   (f32 out)
__global__ __launch_bounds__(128)
void oproj_kernel(const unsigned short* __restrict__ A, const float* __restrict__ W,
                  const float* __restrict__ bias, const float* __restrict__ X,
                  float* __restrict__ Y)
{
  __shared__ unsigned short lA[64][40];   // 80B rows (16B-aligned for async b128)
  __shared__ unsigned short lB[64][34];

  const int tid  = threadIdx.x;
  const int wave = tid >> 5;
  const int lane = tid & 31;
  const int half = lane >> 4;
  const int ln   = lane & 15;
  const int mBase = blockIdx.x * 64;
  const int nBase = blockIdx.y * 64;

  v8f acc[4];
#pragma unroll
  for (int t = 0; t < 4; ++t) { FragF z; for (int r = 0; r < 8; ++r) z.f[r] = 0.f; acc[t] = z.v; }

  const int lr = tid >> 1;
  const int lc = (tid & 1) * 16;

  for (int k0 = 0; k0 < Dc; k0 += 32) {
    __syncthreads();
    {
      // A tile is already bf16 in memory: async global->LDS copy (no VGPR round-trip)
      const unsigned short* sa = A + (size_t)(mBase + lr) * Dc + k0 + lc;
      async_copy32(lds_off_u32(&lA[lr][lc]), sa);
      // W tile needs f32->bf16 conversion: VALU path
      const float* sb = W + (size_t)(nBase + lr) * Dc + k0 + lc;
      unsigned* db = (unsigned*)&lB[lr][lc];
#pragma unroll
      for (int j = 0; j < 8; ++j) db[j] = pack_bf16(sb[2*j], sb[2*j+1]);
    }
    wait_async0();
    __syncthreads();

    v16bf afrag = ldsA_frag(&lA[wave*16 + ln][0], half);
#pragma unroll
    for (int t = 0; t < 4; ++t) {
      v16bf bfrag = ldsB_frag(&lB[t*16 + ln][0], half);
      acc[t] = wmma_bf16(afrag, bfrag, acc[t]);
    }
  }

#pragma unroll
  for (int t = 0; t < 4; ++t) {
    FragF cv; cv.v = acc[t];
    const int col = nBase + t*16 + ln;
    const float bv = bias[col];
#pragma unroll
    for (int r = 0; r < 8; ++r) {
      const int row = mBase + wave*16 + half*8 + r;
      Y[(size_t)row * Dc + col] = cv.f[r] + bv + X[(size_t)row * Dc + col];
    }
  }
}

// ================================================================ LayerNorm
__global__ __launch_bounds__(256)
void ln_kernel(const float* __restrict__ Y, const float* __restrict__ gamma,
               const float* __restrict__ beta, float* __restrict__ Out)
{
  const int row  = blockIdx.x * 8 + (threadIdx.x >> 5);
  const int lane = threadIdx.x & 31;
  const float* y = Y + (size_t)row * Dc;
  float s = 0.f, s2 = 0.f;
#pragma unroll 4
  for (int i = lane; i < Dc; i += 32) { const float v = y[i]; s += v; s2 += v * v; }
#pragma unroll
  for (int off = 1; off < 32; off <<= 1) { s += __shfl_xor(s, off); s2 += __shfl_xor(s2, off); }
  const float mu  = s  * (1.f / Dc);
  const float var = s2 * (1.f / Dc) - mu * mu;
  const float inv = rsqrtf(var + 1e-5f);
  float* o = Out + (size_t)row * Dc;
#pragma unroll 4
  for (int i = lane; i < Dc; i += 32) o[i] = (y[i] - mu) * inv * gamma[i] + beta[i];
}

// ================================================================ launcher
extern "C" void kernel_launch(void* const* d_in, const int* in_sizes, int n_in,
                              void* d_out, int out_size, void* d_ws, size_t ws_size,
                              hipStream_t stream)
{
  (void)in_sizes; (void)n_in; (void)out_size; (void)ws_size;
  const float* x    = (const float*)d_in[0];
  const float* mask = (const float*)d_in[1];
  const float* Wq   = (const float*)d_in[2];
  const float* bq   = (const float*)d_in[3];
  const float* Wk   = (const float*)d_in[4];
  const float* bk   = (const float*)d_in[5];
  const float* Wv   = (const float*)d_in[6];
  const float* bv   = (const float*)d_in[7];
  const float* Wo   = (const float*)d_in[8];
  const float* bo   = (const float*)d_in[9];
  const float* gamma= (const float*)d_in[10];
  const float* beta = (const float*)d_in[11];
  const float* cw   = (const float*)d_in[12];

  const size_t N = (size_t)BS * Dc;
  unsigned short* qb = (unsigned short*)d_ws;      // bf16 Q (post-RoPE)
  unsigned short* kb = qb + N;                     // bf16 K (post-RoPE)
  unsigned short* vb = kb + N;                     // bf16 V
  unsigned short* ob = vb + N;                     // bf16 attn output
  float*          yb = (float*)(ob + N);           // f32 pre-LN

  dim3 gp(BS / 64, Dc / 64, 1);
  proj_kernel<<<gp, 128, 0, stream>>>(x, Wq, bq, qb, 1);
  proj_kernel<<<gp, 128, 0, stream>>>(x, Wk, bk, kb, 1);
  proj_kernel<<<gp, 128, 0, stream>>>(x, Wv, bv, vb, 0);

  dim3 ga(Sc / 64, Hc, Bc);
  attn_kernel<<<ga, 128, 0, stream>>>(qb, kb, vb, mask, cw, ob);

  oproj_kernel<<<gp, 128, 0, stream>>>(ob, Wo, bo, x, yb);
  ln_kernel<<<BS / 8, 256, 0, stream>>>(yb, gamma, beta, (float*)d_out);
}
